// Decoder_40664750359331
// MI455X (gfx1250) — matrix operand order repaired and checked by reading, not compiled, not run on hardware
//
#include <hip/hip_runtime.h>
#include <hip/hip_bf16.h>
#include <math.h>

// Problem constants (from reference)
constexpr int cB = 32, cS = 400, cT = 32, cH = 512, cE = 128, cV = 50000, cA = 512;

typedef __bf16 bf16;
typedef __attribute__((ext_vector_type(16))) __bf16 v16bf;
typedef __attribute__((ext_vector_type(8)))  float v8f;

// Max K for which the shared-A staging path is used (LDS = 32*640*2 = 40 KB)
constexpr int cKmaxLds = 640;

// ---------------------------------------------------------------------------
// WMMA fragment loaders (bf16, 16x16x32). A is row-major [M][K]; B is stored
// transposed [N][K] so each lane reads contiguous K-runs.
// A-layout (ISA 7.12.2, 16-bit A 16x32): lane m (0-15), half=lane>>4:
//   elems 0..7  -> K = 8*half + 0..7
//   elems 8..15 -> K = 8*half + 16..23
// B-layout (K across element index, N across lanes): lane n, half:
//   elems 0..7  -> K = 8*half + 0..7
//   elems 8..15 -> K = 8*half + 16..23
// Templated on the A pointer type so the LDS path keeps addrspace(3) and
// lowers to ds_load instead of flat/global.
// ---------------------------------------------------------------------------
template <typename AP>
__device__ __forceinline__ v16bf load_a_frag(AP A, int row0, int k0, int ld) {
  int lane = threadIdx.x & 31;
  int m = lane & 15, half = lane >> 4;
  auto p = A + (size_t)(row0 + m) * ld + k0 + 8 * half;
  v16bf a;
#pragma unroll
  for (int j = 0; j < 8; ++j) { a[j] = p[j]; a[j + 8] = p[j + 16]; }
  return a;
}

__device__ __forceinline__ v16bf load_b_frag(const bf16* Bt, int col0, int k0, int ld) {
  int lane = threadIdx.x & 31;
  int n = lane & 15, half = lane >> 4;
  const bf16* p = Bt + (size_t)(col0 + n) * ld + k0 + 8 * half;
  v16bf b;
#pragma unroll
  for (int j = 0; j < 8; ++j) { b[j] = p[j]; b[j + 8] = p[j + 16]; }
  return b;
}

// Software-pipelined 32x16 tile core: next k-step's fragments are loaded
// before the current WMMAs issue, so global_load/ds_load latency is hidden
// under the matrix ops instead of a full s_wait_loadcnt 0 per WMMA pair.
template <typename AP>
__device__ __forceinline__ void gemm_core(AP A, const bf16* __restrict__ Bt,
                                          int row0, int col0, int K,
                                          v8f& acc0, v8f& acc1) {
  v16bf b  = load_b_frag(Bt, col0, 0, K);
  v16bf a0 = load_a_frag(A, row0,      0, K);
  v16bf a1 = load_a_frag(A, row0 + 16, 0, K);
  for (int kn = 32; kn < K; kn += 32) {
    __builtin_prefetch(Bt + (size_t)(col0 + (threadIdx.x & 15)) * K + kn + 16, 0, 3);
    v16bf bn  = load_b_frag(Bt, col0, kn, K);
    v16bf a0n = load_a_frag(A, row0,      kn, K);
    v16bf a1n = load_a_frag(A, row0 + 16, kn, K);
    acc0 = __builtin_amdgcn_wmma_f32_16x16x32_bf16(false, a0, false, b, (short)0, acc0,
                                                   false, false);
    acc1 = __builtin_amdgcn_wmma_f32_16x16x32_bf16(false, a1, false, b, (short)0, acc1,
                                                   false, false);
    b = bn; a0 = a0n; a1 = a1n;
  }
  acc0 = __builtin_amdgcn_wmma_f32_16x16x32_bf16(false, a0, false, b, (short)0, acc0,
                                                 false, false);
  acc1 = __builtin_amdgcn_wmma_f32_16x16x32_bf16(false, a1, false, b, (short)0, acc1,
                                                 false, false);
}

// Generic WMMA GEMM: C[M][N] (fp32, row stride ldc) = A[M][K] * Bt[N][K]^T + bias
// Wave tile = 32 rows x 16 cols (one B fragment feeds two WMMAs -> B stream
// traffic halved). When M==32 (mt==1) every wave in the block reads the SAME
// 32xK A matrix, so it is staged once per block into LDS (K<=640 -> 40 KB).
// Requires M%32==0, N%16==0, K%32==0 (true for all shapes here).
// epi: 0 = none, 1 = tanh
__global__ __launch_bounds__(256) void k_gemm_bf16(
    const bf16* __restrict__ A, const bf16* __restrict__ Bt,
    const float* __restrict__ bias, float* __restrict__ C,
    int M, int N, int K, long long ldc, int epi)
{
  __shared__ bf16 sA[32 * cKmaxLds];
  int wave = (int)((blockIdx.x * blockDim.x + threadIdx.x) >> 5);
  int mt = M >> 5, nt = N >> 4;
  bool useLds = (mt == 1) && (K <= cKmaxLds);   // block-uniform
  if (useLds) {
    const uint4* src = (const uint4*)A;         // 8 bf16 per chunk
    uint4* dst = (uint4*)sA;
    int chunks = (32 * K) >> 3;
    for (int i = threadIdx.x; i < chunks; i += 256) dst[i] = src[i];
    __syncthreads();
  }
  if (wave >= mt * nt) return;                   // wave-uniform: EXEC all-1s for WMMA
  int m_tile = wave % mt, n_tile = wave / mt;
  int row0 = m_tile * 32, col0 = n_tile * 16;

  v8f acc0 = {}, acc1 = {};
  if (useLds) gemm_core(sA, Bt, row0, col0, K, acc0, acc1);
  else        gemm_core(A,  Bt, row0, col0, K, acc0, acc1);

  int lane = threadIdx.x & 31;
  int nloc = lane & 15, half = lane >> 4;
  int col = col0 + nloc;
  float bv = bias ? bias[col] : 0.0f;
#pragma unroll
  for (int r = 0; r < 8; ++r) {
    int row = row0 + r + 8 * half;       // C layout: lanes 0-15 -> M=r, 16-31 -> M=r+8
    float v0 = acc0[r] + bv;
    float v1 = acc1[r] + bv;
    if (epi == 1) { v0 = tanhf(v0); v1 = tanhf(v1); }
    C[(size_t)row * (size_t)ldc + col] = v0;
    C[(size_t)(row + 16) * (size_t)ldc + col] = v1;
  }
}

// ------------------------- conversion / pack kernels -----------------------
__global__ void k_cvt(const float* __restrict__ in, bf16* __restrict__ out, int n) {
  int i = blockIdx.x * blockDim.x + threadIdx.x;
  if (i < n) out[i] = (bf16)in[i];
}
// in: [R][C] row-major fp32 -> out: [C][R] row-major bf16
__global__ void k_cvt_t(const float* __restrict__ in, bf16* __restrict__ out, int R, int C) {
  int i = blockIdx.x * blockDim.x + threadIdx.x;
  if (i >= R * C) return;
  int c = i / R, r = i % R;
  out[i] = (bf16)in[(size_t)r * C + c];
}
// Wlstm[n][k] = k<E ? W_ih[n][k] : W_hh[n][k-E]   (n < 4H, k < E+H)
__global__ void k_cat_lstm(const float* __restrict__ W_ih, const float* __restrict__ W_hh,
                           bf16* __restrict__ out) {
  int i = blockIdx.x * blockDim.x + threadIdx.x;
  if (i >= 4 * cH * (cE + cH)) return;
  int n = i / (cE + cH), k = i % (cE + cH);
  float v = (k < cE) ? W_ih[(size_t)n * cE + k] : W_hh[(size_t)n * cH + (k - cE)];
  out[i] = (bf16)v;
}
__global__ void k_bias_sum(const float* a, const float* b, float* out, int n) {
  int i = blockIdx.x * blockDim.x + threadIdx.x;
  if (i < n) out[i] = a[i] + b[i];
}
__global__ void k_init_state(const float* h0, const float* c0, float* h, float* c, float* cov) {
  int i = blockIdx.x * blockDim.x + threadIdx.x;
  if (i < cB * cH) { h[i] = h0[i]; c[i] = c0[i]; }
  if (i < cB * cS) cov[i] = 0.0f;
}
// act[b][k] = k<E ? emb[b,t,k] : h[b][k-E]  (bf16)
__global__ void k_pack_lstm(const float* __restrict__ emb_tgt, const float* __restrict__ h,
                            bf16* __restrict__ act, int t) {
  int i = blockIdx.x * blockDim.x + threadIdx.x;
  if (i >= cB * (cE + cH)) return;
  int b = i / (cE + cH), k = i % (cE + cH);
  float v = (k < cE) ? emb_tgt[((size_t)b * cT + t) * cE + k] : h[b * cH + (k - cE)];
  act[i] = (bf16)v;
}
__global__ void k_pack_ch(const float* __restrict__ ctx, const float* __restrict__ h,
                          bf16* __restrict__ out) {
  int i = blockIdx.x * blockDim.x + threadIdx.x;
  if (i >= cB * 3 * cH) return;
  int b = i / (3 * cH), k = i % (3 * cH);
  float v = (k < 2 * cH) ? ctx[b * 2 * cH + k] : h[b * cH + (k - 2 * cH)];
  out[i] = (bf16)v;
}

// --------------------------- per-step dense math ---------------------------
__device__ __forceinline__ float sigm(float x) { return 1.0f / (1.0f + __expf(-x)); }

__global__ void k_lstm_update(const float* __restrict__ gates, float* h, float* c) {
  int i = blockIdx.x * blockDim.x + threadIdx.x;
  if (i >= cB * cH) return;
  int b = i / cH, j = i % cH;
  const float* g = gates + (size_t)b * 4 * cH;
  float ig = g[j], fg = g[cH + j], gg = g[2 * cH + j], og = g[3 * cH + j];
  float cn = sigm(fg) * c[i] + sigm(ig) * tanhf(gg);
  c[i] = cn;
  h[i] = sigm(og) * tanhf(cn);
}

// scores[b,s] = mask ? sum_a v_att[a]*tanh(enc_proj[b,s,a] + dec_proj[b,a] + cov[b,s]*w_cov[a]) : -1e9
// (b_att already folded into enc_proj). One wave per (b,s).
__global__ __launch_bounds__(256) void k_scores(
    const float* __restrict__ enc_proj, const float* __restrict__ dec_proj,
    const float* __restrict__ cov, const float* __restrict__ w_cov,
    const float* __restrict__ v_att, const unsigned char* __restrict__ mask,
    float* __restrict__ scores)
{
  int gw = (int)((blockIdx.x * blockDim.x + threadIdx.x) >> 5);
  if (gw >= cB * cS) return;
  int b = gw / cS;
  int lane = threadIdx.x & 31;
  float cv = cov[gw];
  const float* ep = enc_proj + (size_t)gw * cA;
  const float* dp = dec_proj + (size_t)b * cA;
  float acc = 0.0f;
  for (int a = lane; a < cA; a += 32)
    acc += v_att[a] * tanhf(ep[a] + dp[a] + cv * w_cov[a]);
#pragma unroll
  for (int off = 16; off; off >>= 1) acc += __shfl_down(acc, off, 32);
  if (lane == 0) scores[gw] = mask[gw] ? acc : -1e9f;
}

__global__ __launch_bounds__(256) void k_softmax_attn(
    const float* __restrict__ scores, float* __restrict__ attn, float* __restrict__ cov,
    float* __restrict__ out_attn, float* __restrict__ out_cov, int t)
{
  __shared__ float sm[256];
  int b = blockIdx.x, tid = threadIdx.x;
  const float* sc = scores + (size_t)b * cS;
  float mx = -3.4e38f;
  for (int s = tid; s < cS; s += 256) mx = fmaxf(mx, sc[s]);
  sm[tid] = mx; __syncthreads();
  for (int k = 128; k; k >>= 1) { if (tid < k) sm[tid] = fmaxf(sm[tid], sm[tid + k]); __syncthreads(); }
  mx = sm[0]; __syncthreads();
  float sum = 0.0f;
  for (int s = tid; s < cS; s += 256) { float e = __expf(sc[s] - mx); attn[b * cS + s] = e; sum += e; }
  sm[tid] = sum; __syncthreads();
  for (int k = 128; k; k >>= 1) { if (tid < k) sm[tid] += sm[tid + k]; __syncthreads(); }
  float inv = 1.0f / sm[0];
  for (int s = tid; s < cS; s += 256) {
    float a = attn[b * cS + s] * inv;
    attn[b * cS + s] = a;
    float cv = cov[b * cS + s] + a;
    cov[b * cS + s] = cv;
    out_attn[((size_t)b * cT + t) * cS + s] = a;
    out_cov[((size_t)b * cT + t) * cS + s]  = cv;
  }
}

__global__ void k_context(const float* __restrict__ attn, const float* __restrict__ enc,
                          float* __restrict__ ctx) {
  int i = blockIdx.x * blockDim.x + threadIdx.x;
  if (i >= cB * 2 * cH) return;
  int b = i / (2 * cH), d = i % (2 * cH);
  const float* e = enc + (size_t)b * cS * (2 * cH) + d;
  const float* a = attn + (size_t)b * cS;
  float acc = 0.0f;
  for (int s = 0; s < cS; ++s) acc += a[s] * e[(size_t)s * (2 * cH)];
  ctx[i] = acc;
}

__global__ void k_ah_out(const float* __restrict__ ah, bf16* __restrict__ ah_bf,
                         float* __restrict__ out_dec, int t) {
  int i = blockIdx.x * blockDim.x + threadIdx.x;
  if (i >= cB * cH) return;
  int b = i / cH, n = i % cH;
  float v = ah[i];
  ah_bf[i] = (bf16)v;
  out_dec[((size_t)b * cT + t) * cH + n] = v;
}

__global__ __launch_bounds__(256) void k_pgen(
    const float* __restrict__ ctx, const float* __restrict__ h,
    const float* __restrict__ emb_tgt, const float* __restrict__ W_pg,
    const float* __restrict__ b_pg, float* __restrict__ pgen,
    float* __restrict__ out_pg, int t)
{
  __shared__ float sm[256];
  int b = blockIdx.x, tid = threadIdx.x;
  float acc = 0.0f;
  for (int k = tid; k < 3 * cH + cE; k += 256) {
    float x;
    if (k < 2 * cH)       x = ctx[b * 2 * cH + k];
    else if (k < 3 * cH)  x = h[b * cH + (k - 2 * cH)];
    else                  x = emb_tgt[((size_t)b * cT + t) * cE + (k - 3 * cH)];
    acc += x * W_pg[k];
  }
  sm[tid] = acc; __syncthreads();
  for (int k = 128; k; k >>= 1) { if (tid < k) sm[tid] += sm[tid + k]; __syncthreads(); }
  if (tid == 0) {
    float p = sigm(sm[0] + b_pg[0]);
    pgen[b] = p;
    out_pg[b * cT + t] = p;
  }
}

__global__ __launch_bounds__(256) void k_rowreduce(
    const float* __restrict__ logp, float* __restrict__ rowmax,
    float* __restrict__ rowsum, int t)
{
  __shared__ float sm[256];
  int b = blockIdx.x, tid = threadIdx.x;
  const float* row = logp + ((size_t)b * cT + t) * (size_t)cV;
  float mx = -3.4e38f;
  for (int v = tid; v < cV; v += 256) mx = fmaxf(mx, row[v]);
  sm[tid] = mx; __syncthreads();
  for (int k = 128; k; k >>= 1) { if (tid < k) sm[tid] = fmaxf(sm[tid], sm[tid + k]); __syncthreads(); }
  mx = sm[0]; __syncthreads();
  float sum = 0.0f;
  for (int v = tid; v < cV; v += 256) sum += __expf(row[v] - mx);
  sm[tid] = sum; __syncthreads();
  for (int k = 128; k; k >>= 1) { if (tid < k) sm[tid] += sm[tid + k]; __syncthreads(); }
  if (tid == 0) { rowmax[b] = mx; rowsum[b] = sm[0]; }
}

__global__ void k_vfinal(float* __restrict__ logp, const float* __restrict__ pgen,
                         const float* __restrict__ rowmax, const float* __restrict__ rowsum, int t) {
  int i = blockIdx.x * blockDim.x + threadIdx.x;
  if (i >= cB * cV) return;
  int b = i / cV, v = i % cV;
  float* row = logp + ((size_t)b * cT + t) * (size_t)cV;
  row[v] = pgen[b] * __expf(row[v] - rowmax[b]) / rowsum[b];
}

__global__ void k_scatter(float* __restrict__ logp, const int* __restrict__ tok,
                          const float* __restrict__ attn, const float* __restrict__ pgen, int t) {
  int i = blockIdx.x * blockDim.x + threadIdx.x;
  if (i >= cB * cS) return;
  int b = i / cS;
  int v = tok[i];
  atomicAdd(logp + ((size_t)b * cT + t) * (size_t)cV + v, (1.0f - pgen[b]) * attn[i]);
}

__global__ void k_log(float* __restrict__ logp, int t) {
  int i = blockIdx.x * blockDim.x + threadIdx.x;
  if (i >= cB * cV) return;
  int b = i / cV, v = i % cV;
  float* row = logp + ((size_t)b * cT + t) * (size_t)cV;
  row[v] = __logf(row[v] + 1e-20f);
}

// ---------------------------------------------------------------------------
static inline size_t align256(size_t x) { return (x + 255) & ~(size_t)255; }
static inline int nblk(long long n, int bs) { return (int)((n + bs - 1) / bs); }

extern "C" void kernel_launch(void* const* d_in, const int* in_sizes, int n_in,
                              void* d_out, int out_size, void* d_ws, size_t ws_size,
                              hipStream_t stream) {
  (void)in_sizes; (void)n_in; (void)out_size; (void)ws_size;
  const int*   src_tok = (const int*)d_in[0];
  const float* emb_tgt = (const float*)d_in[1];
  const float* enc     = (const float*)d_in[2];
  const unsigned char* mask = (const unsigned char*)d_in[3];
  const float* h0   = (const float*)d_in[4];
  const float* c0   = (const float*)d_in[5];
  const float* W_ih = (const float*)d_in[6];
  const float* W_hh = (const float*)d_in[7];
  const float* b_ih = (const float*)d_in[8];
  const float* b_hh = (const float*)d_in[9];
  const float* W_enc = (const float*)d_in[10];
  const float* W_dec = (const float*)d_in[11];
  const float* w_cov = (const float*)d_in[12];
  const float* b_att = (const float*)d_in[13];
  const float* v_att = (const float*)d_in[14];
  const float* W_ah  = (const float*)d_in[15];
  const float* b_ah  = (const float*)d_in[16];
  const float* W_pg  = (const float*)d_in[17];
  const float* b_pg  = (const float*)d_in[18];
  const float* W_v   = (const float*)d_in[19];
  const float* b_v   = (const float*)d_in[20];

  float* out      = (float*)d_out;
  float* out_logp = out;
  float* out_dec  = out_logp + (size_t)cB * cT * cV;
  float* out_attn = out_dec  + (size_t)cB * cT * cH;
  float* out_cov  = out_attn + (size_t)cB * cT * cS;
  float* out_pg   = out_cov  + (size_t)cB * cT * cS;

  // ---- workspace carve-out (~110 MB) ----
  char* w = (char*)d_ws; size_t off = 0;
  auto alloc = [&](size_t bytes) -> void* { void* p = w + off; off = align256(off + bytes); return p; };
  bf16* Wv_t   = (bf16*)alloc((size_t)cV * cH * 2);            // W_v^T  [V][H]
  bf16* Wlstm  = (bf16*)alloc((size_t)4 * cH * (cE + cH) * 2); // [4H][E+H]
  float* b_lstm = (float*)alloc(4 * cH * 4);
  bf16* Wenc_t = (bf16*)alloc((size_t)cA * 2 * cH * 2);        // [A][2H]
  bf16* Wdec_t = (bf16*)alloc((size_t)cA * cH * 2);            // [A][H]
  bf16* Wah_t  = (bf16*)alloc((size_t)cH * 3 * cH * 2);        // [H][3H]
  bf16* enc_bf = (bf16*)alloc((size_t)cB * cS * 2 * cH * 2);   // [B*S][2H]
  float* enc_proj = (float*)alloc((size_t)cB * cS * cA * 4);   // [B*S][A]  (+ b_att)
  float* h     = (float*)alloc(cB * cH * 4);
  float* c     = (float*)alloc(cB * cH * 4);
  float* cov   = (float*)alloc(cB * cS * 4);
  bf16* act_l  = (bf16*)alloc((size_t)cB * (cE + cH) * 2);
  float* gates = (float*)alloc((size_t)cB * 4 * cH * 4);
  bf16* h_bf   = (bf16*)alloc(cB * cH * 2);
  float* dproj = (float*)alloc(cB * cA * 4);
  float* scores = (float*)alloc(cB * cS * 4);
  float* attn  = (float*)alloc(cB * cS * 4);
  float* ctx   = (float*)alloc(cB * 2 * cH * 4);
  bf16* cat_ch = (bf16*)alloc((size_t)cB * 3 * cH * 2);
  float* ah    = (float*)alloc(cB * cH * 4);
  bf16* ah_bf  = (bf16*)alloc(cB * cH * 2);
  float* pgen  = (float*)alloc(cB * 4);
  float* rowmax = (float*)alloc(cB * 4);
  float* rowsum = (float*)alloc(cB * 4);

  auto gemm = [&](const bf16* A, const bf16* Bt, const float* bias, float* C,
                  int M, int N, int K, long long ldc, int epi) {
    long long waves = (long long)(M / 32) * (N / 16);
    k_gemm_bf16<<<nblk(waves * 32, 256), 256, 0, stream>>>(A, Bt, bias, C, M, N, K, ldc, epi);
  };

  // ---- one-time weight conversion / transposition (re-done each call: deterministic) ----
  k_cvt_t<<<nblk((long long)cH * cV, 256), 256, 0, stream>>>(W_v, Wv_t, cH, cV);
  k_cat_lstm<<<nblk((long long)4 * cH * (cE + cH), 256), 256, 0, stream>>>(W_ih, W_hh, Wlstm);
  k_bias_sum<<<nblk(4 * cH, 256), 256, 0, stream>>>(b_ih, b_hh, b_lstm, 4 * cH);
  k_cvt_t<<<nblk(2 * cH * cA, 256), 256, 0, stream>>>(W_enc, Wenc_t, 2 * cH, cA);
  k_cvt_t<<<nblk(cH * cA, 256), 256, 0, stream>>>(W_dec, Wdec_t, cH, cA);
  k_cvt_t<<<nblk(3 * cH * cH, 256), 256, 0, stream>>>(W_ah, Wah_t, 3 * cH, cH);
  k_cvt<<<nblk((long long)cB * cS * 2 * cH, 256), 256, 0, stream>>>(enc, enc_bf, cB * cS * 2 * cH);
  k_init_state<<<nblk(cB * cH, 256), 256, 0, stream>>>(h0, c0, h, c, cov);

  // enc_proj = encoder_states @ W_enc + b_att  : (B*S) x A, K = 2H
  gemm(enc_bf, Wenc_t, b_att, enc_proj, cB * cS, cA, 2 * cH, cA, 0);

  // ---- decode loop (sequential dependency through h, c, cov) ----
  for (int t = 0; t < cT; ++t) {
    // LSTM cell
    k_pack_lstm<<<nblk(cB * (cE + cH), 256), 256, 0, stream>>>(emb_tgt, h, act_l, t);
    gemm(act_l, Wlstm, b_lstm, gates, cB, 4 * cH, cE + cH, 4 * cH, 0);
    k_lstm_update<<<nblk(cB * cH, 256), 256, 0, stream>>>(gates, h, c);

    // coverage attention
    k_cvt<<<nblk(cB * cH, 256), 256, 0, stream>>>(h, h_bf, cB * cH);
    gemm(h_bf, Wdec_t, nullptr, dproj, cB, cA, cH, cA, 0);
    k_scores<<<nblk((long long)cB * cS * 32, 256), 256, 0, stream>>>(
        enc_proj, dproj, cov, w_cov, v_att, mask, scores);
    k_softmax_attn<<<cB, 256, 0, stream>>>(scores, attn, cov, out_attn, out_cov, t);
    k_context<<<nblk(cB * 2 * cH, 256), 256, 0, stream>>>(attn, enc, ctx);

    // attentional hidden state + p_gen
    k_pack_ch<<<nblk(cB * 3 * cH, 256), 256, 0, stream>>>(ctx, h, cat_ch);
    gemm(cat_ch, Wah_t, b_ah, ah, cB, cH, 3 * cH, cH, 1 /*tanh*/);
    k_ah_out<<<nblk(cB * cH, 256), 256, 0, stream>>>(ah, ah_bf, out_dec, t);
    k_pgen<<<cB, 256, 0, stream>>>(ctx, h, emb_tgt, W_pg, b_pg, pgen, out_pg, t);

    // vocab projection (dominant GEMM) directly into logps[:, t, :]
    gemm(ah_bf, Wv_t, b_v, out_logp + (size_t)t * cV, cB, cV, cH, (long long)cT * cV, 0);
    k_rowreduce<<<cB, 256, 0, stream>>>(out_logp, rowmax, rowsum, t);
    k_vfinal<<<nblk((long long)cB * cV, 256), 256, 0, stream>>>(out_logp, pgen, rowmax, rowsum, t);
    k_scatter<<<nblk(cB * cS, 256), 256, 0, stream>>>(out_logp, src_tok, attn, pgen, t);
    k_log<<<nblk((long long)cB * cV, 256), 256, 0, stream>>>(out_logp, t);
  }
}
